// FlashAttentionVarlen_38457137169053
// MI455X (gfx1250) — matrix operand order repaired and checked by reading, not compiled, not run on hardware
//
#include <hip/hip_runtime.h>

// ---------------------------------------------------------------------------
// Varlen flash attention for gfx1250 (CDNA5, wave32, WMMA f16->f32).
// q,k,v: [T, H=16, D=128] fp32 packed; cu_seqlens: [B+1]; out: [T, H, D] fp32.
// scores = softcap(QK^T / sqrt(D)), softmax over valid keys, O = P V.
//
// Per block (128 thr = 4 waves): 64 queries x 1 head x 1 seq.
// Per wave: 16 queries. KV streamed in 32-key chunks through LDS (f16).
//   S^T = K . Q^T   (C layout: lane = query -> softmax nearly lane-local)
//   O^T = V^T . P^T (V staged transposed in LDS; P^T relaid via wave-local LDS)
// ---------------------------------------------------------------------------

typedef __attribute__((ext_vector_type(16))) _Float16 v16h;
typedef __attribute__((ext_vector_type(8)))  _Float16 v8h;
typedef __attribute__((ext_vector_type(4)))  _Float16 v4h;
typedef __attribute__((ext_vector_type(8)))  float    v8f;

#define ATTN_H 16
#define ATTN_D 128
#define KS 136   // Ksh row stride in halfs (128 + 8 pad, keeps 16B align, spreads banks)
#define VS 40    // Vtsh row stride in halfs (32 + 8 pad)
#define PS 40    // Psh row stride in halfs

static constexpr float SOFT_CAP = 20.0f;
static constexpr float NEG_BIG  = -1.0e9f;
static constexpr float QK_SCALE = 0.08838834764831845f; // 1/sqrt(128)

__device__ __forceinline__ float xor16f(float x) {
    // lane <-> lane^16 exchange: ds_swizzle SWAPX16 (and=0x1f, or=0, xor=0x10)
    return __int_as_float(__builtin_amdgcn_ds_swizzle(__float_as_int(x), 0x401F));
}

// CAP * tanh(s * scale / CAP) = CAP - 2*CAP / (1 + e^{s * 2*scale/CAP})
// Exact algebraic rewrite; stable at both tails (e->0 gives -CAP, e->inf gives +CAP).
__device__ __forceinline__ float softcap(float s_raw) {
    const float c1 = QK_SCALE * (2.0f / SOFT_CAP);
    float t = __expf(s_raw * c1);
    return SOFT_CAP - (2.0f * SOFT_CAP) * __builtin_amdgcn_rcpf(1.0f + t);
}

__device__ __forceinline__ v8f wmma_f16(v16h a, v16h b, v8f c) {
    return __builtin_amdgcn_wmma_f32_16x16x32_f16(false, a, false, b, (short)0, c,
                                                  false, false);
}

// Assemble a v16h A/B operand from two contiguous 8-half chunks in LDS.
__device__ __forceinline__ v16h lds16(const _Float16* p0, const _Float16* p1) {
    v8h lo = *(const v8h*)p0;
    v8h hi = *(const v8h*)p1;
    v16h r;
#pragma unroll
    for (int i = 0; i < 8; ++i) { r[i] = lo[i]; r[i + 8] = hi[i]; }
    return r;
}

__global__ __launch_bounds__(128)
void fa_varlen_wmma_kernel(const float* __restrict__ qg,
                           const float* __restrict__ kg,
                           const float* __restrict__ vg,
                           const int* __restrict__ cu_q,
                           const int* __restrict__ cu_k,
                           float* __restrict__ out) {
    __shared__ _Float16 Ksh[32 * KS];          // K tile, [key][d] f16
    __shared__ _Float16 Vtsh[ATTN_D * VS];     // V tile transposed, [d][key] f16
    __shared__ _Float16 Psh[4 * 16 * PS];      // per-wave P^T scratch, [q][key] f16

    const int tid  = threadIdx.x;
    const int lane = tid & 31;
    const int w    = tid >> 5;       // wave id 0..3
    const int g    = lane >> 4;      // half-wave group 0/1
    const int ln   = lane & 15;

    const int b  = blockIdx.z;
    const int h  = blockIdx.y;
    const int qt = blockIdx.x;

    const int qbeg  = cu_q[b];
    const int len_q = cu_q[b + 1] - qbeg;
    const int kbeg  = cu_k[b];
    const int len_k = cu_k[b + 1] - kbeg;

    if (qt * 64 >= len_q) return;

    // ---- Load this wave's 16-query tile into B-layout registers (Q^T) ----
    // B operand (32d x 16q): lane ln = q, element i = d = 32c + 16*g + i.
    const int q_local = qt * 64 + w * 16 + ln;
    const int q_cl    = min(q_local, len_q - 1);
    const size_t qoff = ((size_t)(qbeg + q_cl) * ATTN_H + h) * ATTN_D;

    v16h qB[4];
#pragma unroll
    for (int c = 0; c < 4; ++c) {
        const float4* qp = (const float4*)(qg + qoff + 32 * c + 16 * g);
#pragma unroll
        for (int j = 0; j < 4; ++j) {
            float4 f = qp[j];
            qB[c][4 * j + 0] = (_Float16)f.x;
            qB[c][4 * j + 1] = (_Float16)f.y;
            qB[c][4 * j + 2] = (_Float16)f.z;
            qB[c][4 * j + 3] = (_Float16)f.w;
        }
    }

    v8f acc[8];
#pragma unroll
    for (int dt = 0; dt < 8; ++dt) acc[dt] = (v8f)0.0f;
    float m_run = -1.0e30f;
    float l_run = 0.0f;

    const int pb = w * 16 * PS;   // this wave's Psh base (halfs)

    for (int k0 = 0; k0 < len_k; k0 += 32) {
        // ---- Cooperative stage: 32 keys x 128 d of K and V, fp32 -> f16 ----
#pragma unroll
        for (int it = 0; it < 8; ++it) {
            int e   = tid + it * 128;        // 0..1023 float4 groups
            int row = e >> 5;                // key row 0..31
            int col = (e & 31) * 4;          // d 0..124
            int kl  = min(k0 + row, len_k - 1);
            size_t base = ((size_t)(kbeg + kl) * ATTN_H + h) * ATTN_D + col;
            float4 kf = *(const float4*)(kg + base);
            float4 vf = *(const float4*)(vg + base);
            v4h kh;
            kh[0] = (_Float16)kf.x; kh[1] = (_Float16)kf.y;
            kh[2] = (_Float16)kf.z; kh[3] = (_Float16)kf.w;
            *(v4h*)&Ksh[row * KS + col] = kh;
            Vtsh[(col + 0) * VS + row] = (_Float16)vf.x;
            Vtsh[(col + 1) * VS + row] = (_Float16)vf.y;
            Vtsh[(col + 2) * VS + row] = (_Float16)vf.z;
            Vtsh[(col + 3) * VS + row] = (_Float16)vf.w;
            if (it == 0 && (k0 + row + 32) < len_k) {
                // prefetch next K/V tile -> global_prefetch_b8
                __builtin_prefetch(kg + base + (size_t)32 * ATTN_H * ATTN_D, 0, 1);
                __builtin_prefetch(vg + base + (size_t)32 * ATTN_H * ATTN_D, 0, 1);
            }
        }
        __syncthreads();

        // ---- S^T = K . Q^T for two 16-key subtiles (C: lane=q, rows=keys) ----
        v8f st0 = (v8f)0.0f, st1 = (v8f)0.0f;
#pragma unroll
        for (int c = 0; c < 4; ++c) {
            const int dlo = 32 * c + 16 * g;
            v16h a0 = lds16(&Ksh[ln * KS + dlo], &Ksh[ln * KS + dlo + 8]);
            v16h a1 = lds16(&Ksh[(ln + 16) * KS + dlo], &Ksh[(ln + 16) * KS + dlo + 8]);
            st0 = wmma_f16(a0, qB[c], st0);
            st1 = wmma_f16(a1, qB[c], st1);
        }

        // ---- soft-cap + mask; online softmax (per-lane: one q, 16 keys) ----
        float s[16];
        float tile_max = -1.0e30f;
#pragma unroll
        for (int r = 0; r < 8; ++r) {
            float x0 = softcap(st0[r]);
            float x1 = softcap(st1[r]);
            int key0 = k0 + r + 8 * g;
            int key1 = k0 + 16 + r + 8 * g;
            x0 = (key0 < len_k) ? x0 : NEG_BIG;
            x1 = (key1 < len_k) ? x1 : NEG_BIG;
            s[r]     = x0;
            s[r + 8] = x1;
            tile_max = fmaxf(tile_max, fmaxf(x0, x1));
        }
        tile_max = fmaxf(tile_max, xor16f(tile_max));   // combine lane-pair (same q)
        float m_new = fmaxf(m_run, tile_max);
        float alpha = __expf(m_run - m_new);
        m_run = m_new;

        float rowsum = 0.0f;
#pragma unroll
        for (int r = 0; r < 8; ++r) {
            float p0 = __expf(s[r]     - m_new);
            float p1 = __expf(s[r + 8] - m_new);
            rowsum += p0 + p1;
            // store P^T to per-wave scratch: [q][key_in_chunk]
            Psh[pb + ln * PS + (r + 8 * g)]      = (_Float16)p0;
            Psh[pb + ln * PS + 16 + (r + 8 * g)] = (_Float16)p1;
        }
        l_run = l_run * alpha + rowsum;
#pragma unroll
        for (int dt = 0; dt < 8; ++dt) acc[dt] = acc[dt] * alpha;

        // ---- O^T += V^T . P^T  (A: lane=d row, B: lane=q; same-wave DS order) ----
        v16h pB = lds16(&Psh[pb + ln * PS + 8 * g], &Psh[pb + ln * PS + 16 + 8 * g]);
#pragma unroll
        for (int dt = 0; dt < 8; ++dt) {
            const int d = dt * 16 + ln;
            v16h aV = lds16(&Vtsh[d * VS + 8 * g], &Vtsh[d * VS + 16 + 8 * g]);
            acc[dt] = wmma_f16(aV, pB, acc[dt]);
        }
        __syncthreads();
    }

    // ---- epilogue: normalize and store O (lane=q; 8 contiguous d per dt) ----
    float l_tot = l_run + xor16f(l_run);
    float inv_l = 1.0f / l_tot;

    if (q_local < len_q) {
        const size_t obase = ((size_t)(qbeg + q_local) * ATTN_H + h) * ATTN_D;
#pragma unroll
        for (int dt = 0; dt < 8; ++dt) {
#pragma unroll
            for (int r = 0; r < 8; ++r) {
                out[obase + dt * 16 + 8 * g + r] = acc[dt][r] * inv_l;
            }
        }
    }
}

extern "C" void kernel_launch(void* const* d_in, const int* in_sizes, int n_in,
                              void* d_out, int out_size, void* d_ws, size_t ws_size,
                              hipStream_t stream) {
    (void)n_in; (void)out_size; (void)d_ws; (void)ws_size;
    const float* q  = (const float*)d_in[0];
    const float* k  = (const float*)d_in[1];
    const float* v  = (const float*)d_in[2];
    const int* cu_q = (const int*)d_in[3];
    const int* cu_k = (const int*)d_in[4];
    float* out      = (float*)d_out;

    const int B = in_sizes[3] - 1;                       // cu_seqlens_q has B+1
    const int T = in_sizes[0] / (ATTN_H * ATTN_D);       // packed tokens
    const int QT = (T + 63) / 64;                        // q-tile upper bound

    dim3 grid(QT, ATTN_H, B);
    dim3 block(128, 1, 1);
    fa_varlen_wmma_kernel<<<grid, block, 0, stream>>>(q, k, v, cu_q, cu_k, out);
}
